// MultiheadAttention_50672024158234
// MI455X (gfx1250) — matrix operand order repaired and checked by reading, not compiled, not run on hardware
//
#include <hip/hip_runtime.h>

// ---------------------------------------------------------------------------
// MHA forward for MI455X (gfx1250, wave32, WMMA bf16 16x16x32, f32 accum)
// B=2, N=2048, C=1024, H=16, D=64
// ---------------------------------------------------------------------------

#define Bn 2
#define Nn 2048
#define Cn 1024
#define Hn 16
#define Dn 64
#define TOK (Bn * Nn)          // 4096 rows of x
#define QKVN (3 * Cn)          // 3072

typedef __attribute__((ext_vector_type(8)))  unsigned short u16x8;
typedef __attribute__((ext_vector_type(16))) unsigned short u16x16;
typedef __attribute__((ext_vector_type(16))) __bf16         bf16x16;
typedef __attribute__((ext_vector_type(8)))  float          f32x8;

union FragU { u16x16 u; bf16x16 b; };

__device__ __forceinline__ unsigned short f2bf(float x) {
    unsigned int u = __float_as_uint(x);
    u += 0x7fffu + ((u >> 16) & 1u);          // round-to-nearest-even
    return (unsigned short)(u >> 16);
}

__device__ __forceinline__ bf16x16 make_frag(u16x8 lo, u16x8 hi) {
    FragU f;
#pragma unroll
    for (int i = 0; i < 8; ++i) { f.u[i] = lo[i]; f.u[8 + i] = hi[i]; }
    return f.b;
}

// A fragment: 16x32 bf16 (MxK), row-major source, tile = &A[row0][k0].
// lane holds row m = lane%16; a[i] = A[m][k0 + (lane<16?0:16) + i] (contiguous).
//
__device__ __forceinline__ bf16x16 load_a_frag(const unsigned short* tile, int ld, int lane) {
    int m = lane & 15, hb = lane >> 4;
    const unsigned short* p = tile + m * ld;
    u16x8 lo = *(const u16x8*)(p + hb * 16);
    u16x8 hi = *(const u16x8*)(p + hb * 16 + 8);
    return make_frag(lo, hi);
}

// B fragment: 32x16 bf16 (KxN) taken from a TRANSPOSED (N-major) source,
// tileT = &Bt[col0][k0].  b[i] = B[k0 + (lane<16?0:16) + i][col]  (contiguous).
__device__ __forceinline__ bf16x16 load_b_frag(const unsigned short* tileT, int ld, int lane) {
    int n = lane & 15, hb = lane >> 4;
    const unsigned short* p = tileT + n * ld + hb * 16;
    return make_frag(*(const u16x8*)p, *(const u16x8*)(p + 8));
}

__device__ __forceinline__ f32x8 wmma_bf16(bf16x16 a, bf16x16 b, f32x8 c) {
    return __builtin_amdgcn_wmma_f32_16x16x32_bf16(false, a, false, b, (short)0, c, false, false);
}

// ---------------------------------------------------------------------------
// Kernel 0a: f32 -> bf16 (row-major copy)
// ---------------------------------------------------------------------------
__global__ void k_cvt(const float* __restrict__ src, unsigned short* __restrict__ dst, int n) {
    int i = blockIdx.x * blockDim.x + threadIdx.x;
    if (i < n) dst[i] = f2bf(src[i]);
}

// Kernel 0b: w[K][N] (f32) -> wt[N][K] (bf16), contiguous writes
__global__ void k_cvt_tr(const float* __restrict__ w, unsigned short* __restrict__ wt, int K, int N) {
    int i = blockIdx.x * blockDim.x + threadIdx.x;
    if (i < K * N) {
        int n = i / K, k = i - n * K;
        wt[i] = f2bf(w[k * N + n]);
    }
}

// ---------------------------------------------------------------------------
// Kernel 1: QKV projection.  xh[4096][1024] x wtq[3072][1024]^T -> Q/K/Vt
// WG = 256 thr = 8 waves (2x4), WG tile 128x256, wave tile 64x64
// (4x4 WMMA tiles -> 16 wmma per 16 b128 loads per 32-K step).
// Q: [BH][N][D] bf16 pre-scaled by 1/sqrt(D);  K: [BH][N][D];  Vt: [BH][D][N]
// ---------------------------------------------------------------------------
__global__ void __launch_bounds__(256) k_gemm_qkv(const unsigned short* __restrict__ xh,
                                                  const unsigned short* __restrict__ wtq,
                                                  unsigned short* __restrict__ Qb,
                                                  unsigned short* __restrict__ Kb,
                                                  unsigned short* __restrict__ Vtb) {
    int lane = threadIdx.x & 31, wave = threadIdx.x >> 5;
    int rowBase = blockIdx.x * 128 + (wave & 1) * 64;
    int colBase = blockIdx.y * 256 + (wave >> 1) * 64;

    f32x8 acc[4][4] = {};
    for (int k0 = 0; k0 < Cn; k0 += 32) {
        bf16x16 a[4], b[4];
#pragma unroll
        for (int i = 0; i < 4; ++i)
            a[i] = load_a_frag(xh + (size_t)(rowBase + i * 16) * Cn + k0, Cn, lane);
#pragma unroll
        for (int j = 0; j < 4; ++j)
            b[j] = load_b_frag(wtq + (size_t)(colBase + j * 16) * Cn + k0, Cn, lane);
#pragma unroll
        for (int i = 0; i < 4; ++i)
#pragma unroll
            for (int j = 0; j < 4; ++j)
                acc[i][j] = wmma_bf16(a[i], b[j], acc[i][j]);
    }

    int ln = lane & 15, hb = lane >> 4;
#pragma unroll
    for (int i = 0; i < 4; ++i) {
#pragma unroll
        for (int j = 0; j < 4; ++j) {
            int col = colBase + j * 16 + ln;
            int t = col >> 10, h = (col >> 6) & 15, d = col & 63;
#pragma unroll
            for (int r = 0; r < 8; ++r) {
                int row = rowBase + i * 16 + hb * 8 + r;    // token index in [0,4096)
                int bh = (row >> 11) * Hn + h;
                int n = row & (Nn - 1);
                float v = acc[i][j][r];
                if (t == 0)      Qb[((size_t)bh * Nn + n) * Dn + d] = f2bf(v * 0.125f); // 1/sqrt(64)
                else if (t == 1) Kb[((size_t)bh * Nn + n) * Dn + d] = f2bf(v);
                else             Vtb[((size_t)bh * Dn + d) * Nn + n] = f2bf(v);
            }
        }
    }
}

// ---------------------------------------------------------------------------
// Kernel 2: causal flash attention, transposed tiles.
// Each wave owns 32 queries (two 16-wide q-tiles sharing K/V fragments).
// S^T (k x q) = Kmat x Q^T -> per-lane softmax state.  P staged via LDS,
// O^T += Vt x P^T.  Obuf: [tok][C] bf16.
// ---------------------------------------------------------------------------
__global__ void __launch_bounds__(256) k_attn(const unsigned short* __restrict__ Qb,
                                              const unsigned short* __restrict__ Kb,
                                              const unsigned short* __restrict__ Vtb,
                                              unsigned short* __restrict__ Obuf) {
    __shared__ __align__(16) unsigned short Pls[8 * 1024];  // 2KB per wave
    int lane = threadIdx.x & 31, wave = threadIdx.x >> 5;
    int ln = lane & 15, hb = lane >> 4;
    int bh = blockIdx.y;                    // [0,32)
    int qbase = blockIdx.x * 256 + wave * 32;

    const unsigned short* Qh = Qb + (size_t)bh * Nn * Dn;
    const unsigned short* Kh = Kb + (size_t)bh * Nn * Dn;
    const unsigned short* Vh = Vtb + (size_t)bh * Dn * Nn;
    unsigned short* Pw = Pls + wave * 1024;

    // Q^T B-fragments: 2 q-tiles x 2 d-chunks, loaded once
    bf16x16 qf[2][2];
#pragma unroll
    for (int j = 0; j < 2; ++j)
#pragma unroll
        for (int c = 0; c < 2; ++c)
            qf[j][c] = load_b_frag(Qh + (size_t)(qbase + j * 16) * Dn + c * 32, Dn, lane);

    float m[2] = { -1e30f, -1e30f }, l[2] = { 0.0f, 0.0f };
    f32x8 o[2][4] = {};                     // O^T per q-tile: 4 d-tiles

    int kcMax = (qbase + 31) >> 5;          // 32-k chunks up to the diagonal
    for (int kc = 0; kc <= kcMax; ++kc) {
        // ---- scores: 2 k-tiles x 2 q-tiles of 16x16 S^T ----
        f32x8 s[2][2];                      // [q-tile][k-tile]
#pragma unroll
        for (int t = 0; t < 2; ++t) {
            int kt = kc * 2 + t;
            bf16x16 ka0 = load_a_frag(Kh + (size_t)kt * 16 * Dn, Dn, lane);
            bf16x16 ka1 = load_a_frag(Kh + (size_t)kt * 16 * Dn + 32, Dn, lane);
#pragma unroll
            for (int j = 0; j < 2; ++j) {
                f32x8 sv = {};
                sv = wmma_bf16(ka0, qf[j][0], sv);
                sv = wmma_bf16(ka1, qf[j][1], sv);
                int qg = qbase + j * 16 + ln;
#pragma unroll
                for (int r = 0; r < 8; ++r) {   // causal mask: k > q
                    int kg = kt * 16 + hb * 8 + r;
                    if (kg > qg) sv[r] = -1e30f;
                }
                s[j][t] = sv;
            }
        }
        // ---- online softmax per q-tile + stage P into LDS ----
#pragma unroll
        for (int j = 0; j < 2; ++j) {
            float cm = -1e30f;
#pragma unroll
            for (int t = 0; t < 2; ++t)
#pragma unroll
                for (int r = 0; r < 8; ++r) cm = fmaxf(cm, s[j][t][r]);
            cm = fmaxf(cm, __shfl_xor(cm, 16, 32));
            float nm = fmaxf(m[j], cm);
            float alpha = __expf(m[j] - nm);
            m[j] = nm;
            float rs = 0.0f;
#pragma unroll
            for (int t = 0; t < 2; ++t)
#pragma unroll
                for (int r = 0; r < 8; ++r) {
                    float p = __expf(s[j][t][r] - nm);
                    s[j][t][r] = p;
                    rs += p;
                }
            rs += __shfl_xor(rs, 16, 32);
            l[j] = l[j] * alpha + rs;
#pragma unroll
            for (int dt = 0; dt < 4; ++dt)
#pragma unroll
                for (int r = 0; r < 8; ++r) o[j][dt][r] *= alpha;
            // P[q][k] layout: contiguous 16B per lane
#pragma unroll
            for (int t = 0; t < 2; ++t) {
                u16x8 pv;
#pragma unroll
                for (int r = 0; r < 8; ++r) pv[r] = f2bf(s[j][t][r]);
                *(u16x8*)&Pw[j * 512 + ln * 32 + t * 16 + hb * 8] = pv;
            }
        }
        __builtin_amdgcn_wave_barrier();
        asm volatile("s_wait_dscnt 0" ::: "memory");   // same-wave DS RAW

        // ---- O^T += Vt x P^T, V fragments shared by both q-tiles ----
        bf16x16 vf[4];
#pragma unroll
        for (int dt = 0; dt < 4; ++dt)
            vf[dt] = load_a_frag(Vh + (size_t)(dt * 16) * Nn + kc * 32, Nn, lane);
#pragma unroll
        for (int j = 0; j < 2; ++j) {
            const unsigned short* pp = Pw + j * 512 + ln * 32 + hb * 16;
            bf16x16 pb = make_frag(*(const u16x8*)pp, *(const u16x8*)(pp + 8));
#pragma unroll
            for (int dt = 0; dt < 4; ++dt)
                o[j][dt] = wmma_bf16(vf[dt], pb, o[j][dt]);
        }
        __builtin_amdgcn_wave_barrier();
    }

    // ---- epilogue: O^T / l  ->  Obuf[tok][h*64 + d] (contiguous 16B) ----
    int b = bh >> 4, h = bh & 15;
#pragma unroll
    for (int j = 0; j < 2; ++j) {
        float invl = 1.0f / l[j];
        size_t row = (size_t)b * Nn + qbase + j * 16 + ln;
#pragma unroll
        for (int dt = 0; dt < 4; ++dt) {
            u16x8 ov;
#pragma unroll
            for (int r = 0; r < 8; ++r) ov[r] = f2bf(o[j][dt][r] * invl);
            *(u16x8*)&Obuf[row * Cn + h * Dn + dt * 16 + hb * 8] = ov;
        }
    }
}

// ---------------------------------------------------------------------------
// Kernel 3: output projection + bias.  Obuf x wto^T + b_out -> out (f32)
// Same 64x64 wave tile as kernel 1.
// ---------------------------------------------------------------------------
__global__ void __launch_bounds__(256) k_gemm_out(const unsigned short* __restrict__ Obuf,
                                                  const unsigned short* __restrict__ wto,
                                                  const float* __restrict__ bias,
                                                  float* __restrict__ out) {
    int lane = threadIdx.x & 31, wave = threadIdx.x >> 5;
    int rowBase = blockIdx.x * 128 + (wave & 1) * 64;
    int colBase = blockIdx.y * 256 + (wave >> 1) * 64;

    f32x8 acc[4][4] = {};
    for (int k0 = 0; k0 < Cn; k0 += 32) {
        bf16x16 a[4], b[4];
#pragma unroll
        for (int i = 0; i < 4; ++i)
            a[i] = load_a_frag(Obuf + (size_t)(rowBase + i * 16) * Cn + k0, Cn, lane);
#pragma unroll
        for (int j = 0; j < 4; ++j)
            b[j] = load_b_frag(wto + (size_t)(colBase + j * 16) * Cn + k0, Cn, lane);
#pragma unroll
        for (int i = 0; i < 4; ++i)
#pragma unroll
            for (int j = 0; j < 4; ++j)
                acc[i][j] = wmma_bf16(a[i], b[j], acc[i][j]);
    }

    int ln = lane & 15, hb = lane >> 4;
#pragma unroll
    for (int i = 0; i < 4; ++i) {
#pragma unroll
        for (int j = 0; j < 4; ++j) {
            int col = colBase + j * 16 + ln;
            float bb = bias[col];
#pragma unroll
            for (int r = 0; r < 8; ++r) {
                int row = rowBase + i * 16 + hb * 8 + r;
                out[(size_t)row * Cn + col] = acc[i][j][r] + bb;
            }
        }
    }
}

// ---------------------------------------------------------------------------
// Host launcher.  Workspace layout (48 MiB total):
//   xh 8MiB | wtq 6MiB | wto 2MiB | Q 8MiB | K 8MiB | Vt 8MiB | Obuf 8MiB
// ---------------------------------------------------------------------------
extern "C" void kernel_launch(void* const* d_in, const int* in_sizes, int n_in,
                              void* d_out, int out_size, void* d_ws, size_t ws_size,
                              hipStream_t stream) {
    const float* x     = (const float*)d_in[0];   // [B,N,C]
    const float* w_qkv = (const float*)d_in[1];   // [C,3C]
    const float* w_out = (const float*)d_in[2];   // [C,C]
    const float* b_out = (const float*)d_in[3];   // [C]
    // d_in[4] attn_mask: causal, computed analytically in-kernel.

    char* ws = (char*)d_ws;
    unsigned short* xh  = (unsigned short*)(ws);                         //  8 MiB
    unsigned short* wtq = (unsigned short*)(ws + 8388608);               //  6 MiB
    unsigned short* wto = (unsigned short*)(ws + 14680064);              //  2 MiB
    unsigned short* Qb  = (unsigned short*)(ws + 16777216);              //  8 MiB
    unsigned short* Kb  = (unsigned short*)(ws + 25165824);              //  8 MiB
    unsigned short* Vtb = (unsigned short*)(ws + 33554432);              //  8 MiB
    unsigned short* Ob  = (unsigned short*)(ws + 41943040);              //  8 MiB
    float* out = (float*)d_out;

    // 0) precision conversion / weight transposition
    k_cvt<<<(TOK * Cn + 255) / 256, 256, 0, stream>>>(x, xh, TOK * Cn);
    k_cvt_tr<<<(Cn * QKVN + 255) / 256, 256, 0, stream>>>(w_qkv, wtq, Cn, QKVN);
    k_cvt_tr<<<(Cn * Cn + 255) / 256, 256, 0, stream>>>(w_out, wto, Cn, Cn);

    // 1) QKV projection (M=4096, K=1024, N=3072): 32x12 WGs of 128x256
    k_gemm_qkv<<<dim3(TOK / 128, QKVN / 256), 256, 0, stream>>>(xh, wtq, Qb, Kb, Vtb);

    // 2) causal attention: 8 q-blocks x 32 (b,h) pairs
    k_attn<<<dim3(Nn / 256, Bn * Hn), 256, 0, stream>>>(Qb, Kb, Vtb, Ob);

    // 3) output projection + bias (M=4096, K=1024, N=1024): 32x4 WGs of 128x256
    k_gemm_out<<<dim3(TOK / 128, Cn / 256), 256, 0, stream>>>(Ob, wto, b_out, out);
}
